// Atten_86552180949549
// MI455X (gfx1250) — matrix operand order repaired and checked by reading, not compiled, not run on hardware
//
#include <hip/hip_runtime.h>
#include <hip/hip_bf16.h>

#define B_    8
#define S_    1500
#define D_    768
#define H_    12
#define DH_   64
#define SPAD  1504          // S padded to multiple of 32 for K/V tiles
#define SCALE_ 0.125f

typedef _Float16 half_t;
typedef half_t v16h __attribute__((ext_vector_type(16)));
typedef half_t v8h  __attribute__((ext_vector_type(8)));
typedef half_t v4h  __attribute__((ext_vector_type(4)));
typedef float  v8f  __attribute__((ext_vector_type(8)));

// D = A(16x32) * B(32x16) + C, f32 accum
__device__ __forceinline__ v8f wmma16(v16h a, v16h b, v8f c) {
  return __builtin_amdgcn_wmma_f32_16x16x32_f16(false, a, false, b, (short)0, c,
                                                false, false);
}

// A-fragment (16x32, 16-bit): lane m(0..15) holds row m K 0..15,
// lane m+16 holds row m K 16..31  -> two 16B contiguous chunks.
__device__ __forceinline__ v16h load_a(const half_t* base, int stride,
                                       int row, int k0, int hi) {
  const half_t* p = base + (size_t)row * stride + k0 + (hi ? 16 : 0);
  v8h lo = *(const v8h*)p;
  v8h hp = *(const v8h*)(p + 8);
  v16h r;
#pragma unroll
  for (int i = 0; i < 8; ++i) { r[i] = lo[i]; r[i + 8] = hp[i]; }
  return r;
}

// B-fragment (32x16, 16-bit): lane n holds column n&15, K 0..15 (lanes 0-15)
// or K 16..31 (lanes 16-31)  -> one 32B contiguous chunk.
__device__ __forceinline__ v16h load_b(const half_t* base, int stride,
                                       int col, int k0, int hi) {
  return *(const v16h*)(base + (size_t)col * stride + k0 + (hi ? 16 : 0));
}

// ---------------- f32 -> f16 conversion ----------------
__global__ void cvt_kernel(const float* __restrict__ in, half_t* __restrict__ out,
                           int n4) {
  int i = blockIdx.x * blockDim.x + threadIdx.x;
  if (i < n4) {
    float4 v = ((const float4*)in)[i];
    v4h o;
    o[0] = (half_t)v.x; o[1] = (half_t)v.y; o[2] = (half_t)v.z; o[3] = (half_t)v.w;
    ((v4h*)out)[i] = o;
  }
}

// ---------------- Q/K projection: out^T = W * x^T ----------------
// A = W tile (16 channels x 32 k), B = x^T (32 k x 16 rows).
// D: lane = x-row, VGPR = channel -> contiguous dh store into [B,H,SPAD,64].
__global__ void proj_qk_kernel(const half_t* __restrict__ xh,
                               const half_t* __restrict__ Wh,
                               const float* __restrict__ bias, float scale,
                               half_t* __restrict__ out) {
  const int lane = threadIdx.x & 31;
  const int hi = lane >> 4, l15 = lane & 15;
  const int wave = __builtin_amdgcn_readfirstlane(threadIdx.x >> 5);
  const int wid = blockIdx.x * 8 + wave;          // scalar
  const int mt = wid / 188;                       // channel tile 0..47
  const int ng = wid % 188;                       // group of 4 row-tiles
  if (mt >= 48) return;
  const int c0 = mt * 16;
  int nt[4];
#pragma unroll
  for (int t = 0; t < 4; ++t) {
    int n = ng * 4 + t;
    nt[t] = n < 750 ? n : 749;                    // clamp: compute, don't store
  }
  v8f acc[4] = {};
  for (int k0 = 0; k0 < D_; k0 += 32) {
    v16h a = load_a(Wh, D_, c0 + l15, k0, hi);
    v16h b[4];
#pragma unroll
    for (int t = 0; t < 4; ++t) b[t] = load_b(xh, D_, nt[t] * 16 + l15, k0, hi);
#pragma unroll
    for (int t = 0; t < 4; ++t) acc[t] = wmma16(a, b[t], acc[t]);
  }
  const int cb = c0 + hi * 8;                     // first channel this lane holds
  const int h = cb / DH_, dh = cb % DH_;
  float bv[8];
#pragma unroll
  for (int i = 0; i < 8; ++i) bv[i] = bias ? bias[cb + i] : 0.f;
#pragma unroll
  for (int t = 0; t < 4; ++t) {
    if (ng * 4 + t >= 750) continue;              // scalar branch
    int r = nt[t] * 16 + l15;                     // flat row in [B*S]
    int bb = r / S_, s = r % S_;
    half_t* dst = out + ((((size_t)bb * H_ + h) * SPAD + s) * DH_ + dh);
    v8h o;
#pragma unroll
    for (int i = 0; i < 8; ++i) o[i] = (half_t)((acc[t][i] + bv[i]) * scale);
    *(v8h*)dst = o;
  }
}

// ---------------- V projection -> V^T [B,H,64,SPAD] ----------------
// Normal orientation: A = x tile, B(k,n) = Wv[n,k]. D: lane = channel,
// VGPR = s-row -> contiguous s store into V^T.
__global__ void proj_v_kernel(const half_t* __restrict__ xh,
                              const half_t* __restrict__ Wh,
                              const float* __restrict__ bias,
                              half_t* __restrict__ outT) {
  const int lane = threadIdx.x & 31;
  const int hi = lane >> 4, l15 = lane & 15;
  const int wave = __builtin_amdgcn_readfirstlane(threadIdx.x >> 5);
  const int wid = blockIdx.x * 8 + wave;          // scalar
  if (wid >= 9000) return;
  const int rt = wid / 12;                        // row tile 0..749
  const int cg = wid % 12;                        // 64-channel group
  const int r0 = rt * 16, c0 = cg * 64;
  v8f acc[4] = {};
  for (int k0 = 0; k0 < D_; k0 += 32) {
    v16h a = load_a(xh, D_, r0 + l15, k0, hi);
    v16h b[4];
#pragma unroll
    for (int t = 0; t < 4; ++t) b[t] = load_b(Wh, D_, c0 + t * 16 + l15, k0, hi);
#pragma unroll
    for (int t = 0; t < 4; ++t) acc[t] = wmma16(a, b[t], acc[t]);
  }
#pragma unroll
  for (int t = 0; t < 4; ++t) {
    const int c = c0 + t * 16 + l15;
    const float bv = bias[c];
    const int h = c / DH_, dh = c % DH_;
#pragma unroll
    for (int i = 0; i < 8; ++i) {
      int r = r0 + hi * 8 + i;
      int bb = r / S_, s = r % S_;
      outT[(((size_t)bb * H_ + h) * DH_ + dh) * SPAD + s] = (half_t)(acc[t][i] + bv);
    }
  }
}

// ---------------- Flash attention ----------------
// Block = 8 waves, all on one (b,h); each wave owns one 16-query tile.
// K/V^T tiles for each 32-key step are staged cooperatively in LDS.
// Scores as S^T = K * Q^T (lane = query, VGPR = key) -> per-lane softmax.
// ctx^T = V^T * P^T accumulated over 32-key steps.
#define KSTR 72   // LDS row stride (halfs) for K tile, padded vs 64
#define VSTR 40   // LDS row stride (halfs) for V tile, padded vs 32
__global__ void attn_kernel(const half_t* __restrict__ Qh,
                            const half_t* __restrict__ Kh,
                            const half_t* __restrict__ Vth,
                            half_t* __restrict__ ctx) {
  __shared__ half_t Kl[32 * KSTR];
  __shared__ half_t Vl[64 * VSTR];
  const int tid = threadIdx.x;
  const int lane = tid & 31;
  const int hi = lane >> 4, l15 = lane & 15;
  const int wave = __builtin_amdgcn_readfirstlane(tid >> 5);
  const int bh = blockIdx.x / 12;                 // 96 heads
  const int qg = blockIdx.x % 12;                 // group of 8 q-tiles
  const int qt0 = qg * 8 + wave;                  // 0..95
  const int qt = qt0 < 94 ? qt0 : 93;             // clamp: keep wave collective
  const half_t* Qb = Qh + (size_t)bh * SPAD * DH_;
  const half_t* Kb = Kh + (size_t)bh * SPAD * DH_;
  const half_t* Vb = Vth + (size_t)bh * DH_ * SPAD;
  const int q = qt * 16 + l15;
  const int qc = q < S_ ? q : S_ - 1;
  const v16h bq0 = load_b(Qb, DH_, qc, 0, hi);    // Q^T, d 0..31
  const v16h bq1 = load_b(Qb, DH_, qc, 32, hi);   // Q^T, d 32..63
  // cooperative-stage indices (256 threads, one 16B chunk each per tile)
  const int kr = tid >> 3, kc = (tid & 7) * 8;    // K: 32 rows x 8 chunks
  const int vr = tid >> 2, vc = (tid & 3) * 8;    // V: 64 rows x 4 chunks
  v8f acc[4] = {};
  float m = -__builtin_inff();
  float l = 0.f;
  for (int k0 = 0; k0 < S_; k0 += 32) {
    // stage K (32 keys x 64 d) and V^T (64 d x 32 keys) into LDS
    *(v8h*)&Kl[kr * KSTR + kc] = *(const v8h*)(Kb + (size_t)(k0 + kr) * DH_ + kc);
    *(v8h*)&Vl[vr * VSTR + vc] = *(const v8h*)(Vb + (size_t)vr * SPAD + k0 + vc);
    if (k0 + 32 < S_)
      __builtin_prefetch(Kb + (size_t)(k0 + 32 + kr) * DH_ + kc, 0, 1);
    __syncthreads();
    // scores for 32 keys (two 16x16 S^T tiles)
    v16h ak0l = load_a(Kl, KSTR, l15, 0, hi);
    v16h ak0h = load_a(Kl, KSTR, l15, 32, hi);
    v16h ak1l = load_a(Kl, KSTR, 16 + l15, 0, hi);
    v16h ak1h = load_a(Kl, KSTR, 16 + l15, 32, hi);
    v8f s0 = {}, s1 = {};
    s0 = wmma16(ak0l, bq0, s0);
    s0 = wmma16(ak0h, bq1, s0);
    s1 = wmma16(ak1l, bq0, s1);
    s1 = wmma16(ak1h, bq1, s1);
    float p0[8], p1[8];
    float mloc = -__builtin_inff();
#pragma unroll
    for (int i = 0; i < 8; ++i) {
      const int kk0 = k0 + hi * 8 + i;
      p0[i] = (kk0 < S_) ? s0[i] : -__builtin_inff();
      p1[i] = (kk0 + 16 < S_) ? s1[i] : -__builtin_inff();
      mloc = fmaxf(mloc, fmaxf(p0[i], p1[i]));
    }
    mloc = fmaxf(mloc, __shfl_xor(mloc, 16));     // combine key halves
    const float mn = fmaxf(m, mloc);
    const float alpha = __expf(m - mn);
    m = mn;
    l *= alpha;
#pragma unroll
    for (int t = 0; t < 4; ++t)
#pragma unroll
      for (int i = 0; i < 8; ++i) acc[t][i] *= alpha;
    float ls = 0.f;
#pragma unroll
    for (int i = 0; i < 8; ++i) {
      p0[i] = __expf(p0[i] - mn);
      p1[i] = __expf(p1[i] - mn);
      ls += p0[i] + p1[i];
    }
    l += ls + __shfl_xor(ls, 16);
    // Repack P^T into a B-fragment (keys x queries), f16.
    v16h bp;
#pragma unroll
    for (int i = 0; i < 8; ++i) {
      const float x0 = __shfl_xor(p0[i], 16);
      const float x1 = __shfl_xor(p1[i], 16);
      bp[i]     = (half_t)(hi ? x1 : p0[i]);      // keys k0+16*hi+0..7
      bp[i + 8] = (half_t)(hi ? p1[i] : x0);      // keys k0+16*hi+8..15
    }
    // ctx^T += V^T * P^T (4 d-tiles)
    v16h av[4];
#pragma unroll
    for (int t = 0; t < 4; ++t) av[t] = load_a(Vl, VSTR, t * 16 + l15, 0, hi);
#pragma unroll
    for (int t = 0; t < 4; ++t) acc[t] = wmma16(av[t], bp, acc[t]);
    __syncthreads();                              // before next overwrite
  }
  const float inv = 1.f / l;
  if (qt0 < 94 && q < S_) {
    const int b = bh / H_, h = bh % H_;
    half_t* dst = ctx + ((size_t)(b * S_ + q)) * D_ + h * DH_ + hi * 8;
#pragma unroll
    for (int t = 0; t < 4; ++t) {
      v8h o;
#pragma unroll
      for (int i = 0; i < 8; ++i) o[i] = (half_t)(acc[t][i] * inv);
      *(v8h*)(dst + t * 16) = o;
    }
  }
}

// ---------------- Output projection: out = ctx @ Wo^T + bo (f32) ----------------
__global__ void proj_o_kernel(const half_t* __restrict__ ch,
                              const half_t* __restrict__ Wh,
                              const float* __restrict__ bias,
                              float* __restrict__ out) {
  const int lane = threadIdx.x & 31;
  const int hi = lane >> 4, l15 = lane & 15;
  const int wave = __builtin_amdgcn_readfirstlane(threadIdx.x >> 5);
  const int wid = blockIdx.x * 8 + wave;          // scalar
  const int mt = wid / 188;
  const int ng = wid % 188;
  if (mt >= 48) return;
  const int c0 = mt * 16;
  int nt[4];
#pragma unroll
  for (int t = 0; t < 4; ++t) {
    int n = ng * 4 + t;
    nt[t] = n < 750 ? n : 749;
  }
  v8f acc[4] = {};
  for (int k0 = 0; k0 < D_; k0 += 32) {
    v16h a = load_a(Wh, D_, c0 + l15, k0, hi);
    v16h b[4];
#pragma unroll
    for (int t = 0; t < 4; ++t) b[t] = load_b(ch, D_, nt[t] * 16 + l15, k0, hi);
#pragma unroll
    for (int t = 0; t < 4; ++t) acc[t] = wmma16(a, b[t], acc[t]);
  }
  const int cb = c0 + hi * 8;
  float bv[8];
#pragma unroll
  for (int i = 0; i < 8; ++i) bv[i] = bias[cb + i];
#pragma unroll
  for (int t = 0; t < 4; ++t) {
    if (ng * 4 + t >= 750) continue;              // scalar branch
    int r = nt[t] * 16 + l15;
    v8f o;
#pragma unroll
    for (int i = 0; i < 8; ++i) o[i] = acc[t][i] + bv[i];
    *(v8f*)(out + (size_t)r * D_ + cb) = o;
  }
}

static inline size_t alup(size_t x) { return (x + 255) & ~(size_t)255; }

extern "C" void kernel_launch(void* const* d_in, const int* in_sizes, int n_in,
                              void* d_out, int out_size, void* d_ws, size_t ws_size,
                              hipStream_t stream) {
  const float* x  = (const float*)d_in[0];
  const float* Wq = (const float*)d_in[1];
  const float* bq = (const float*)d_in[2];
  const float* Wk = (const float*)d_in[3];
  const float* Wv = (const float*)d_in[4];
  const float* bv = (const float*)d_in[5];
  const float* Wo = (const float*)d_in[6];
  const float* bo = (const float*)d_in[7];
  float* out = (float*)d_out;

  const size_t xh_b = (size_t)B_ * S_ * D_ * sizeof(half_t);
  const size_t w_b  = (size_t)D_ * D_ * sizeof(half_t);
  const size_t qk_b = (size_t)B_ * H_ * SPAD * DH_ * sizeof(half_t);

  char* p = (char*)d_ws;
  half_t* xh   = (half_t*)p; p += alup(xh_b);
  half_t* Wqh  = (half_t*)p; p += alup(w_b);
  half_t* Wkh  = (half_t*)p; p += alup(w_b);
  half_t* Wvh  = (half_t*)p; p += alup(w_b);
  half_t* Woh  = (half_t*)p; p += alup(w_b);
  half_t* Qhb  = (half_t*)p; p += alup(qk_b);
  half_t* Khb  = (half_t*)p; p += alup(qk_b);
  half_t* Vthb = (half_t*)p; p += alup(qk_b);
  half_t* ctxh = (half_t*)p; p += alup(xh_b);
  (void)ws_size; (void)in_sizes; (void)n_in; (void)out_size;

  int n4 = B_ * S_ * D_ / 4;
  cvt_kernel<<<(n4 + 255) / 256, 256, 0, stream>>>(x, xh, n4);
  n4 = D_ * D_ / 4;
  cvt_kernel<<<(n4 + 255) / 256, 256, 0, stream>>>(Wq, Wqh, n4);
  cvt_kernel<<<(n4 + 255) / 256, 256, 0, stream>>>(Wk, Wkh, n4);
  cvt_kernel<<<(n4 + 255) / 256, 256, 0, stream>>>(Wv, Wvh, n4);
  cvt_kernel<<<(n4 + 255) / 256, 256, 0, stream>>>(Wo, Woh, n4);

  // 48 channel-tiles * 188 row-groups = 9024 waves = 1128 blocks of 8 waves
  proj_qk_kernel<<<1128, 256, 0, stream>>>(xh, Wqh, bq, SCALE_, Qhb);
  proj_qk_kernel<<<1128, 256, 0, stream>>>(xh, Wkh, nullptr, 1.0f, Khb);
  // 750 row-tiles * 12 channel-groups = 9000 waves
  proj_v_kernel<<<1125, 256, 0, stream>>>(xh, Wvh, bv, Vthb);
  // 96 heads * 12 q-tile groups = 1152 blocks (8 q-tiles each)
  attn_kernel<<<1152, 256, 0, stream>>>(Qhb, Khb, Vthb, ctxh);
  proj_o_kernel<<<1128, 256, 0, stream>>>(ctxh, Woh, bo, out);
}
